// DeepSeekMoELayer_63788854280365
// MI455X (gfx1250) — matrix operand order repaired and checked by reading, not compiled, not run on hardware
//
#include <hip/hip_runtime.h>

// ---------------------------------------------------------------------------
// MI455X (gfx1250, wave32) MoE layer.  All GEMM FLOPs (~670 GFLOP) go through
// v_wmma_f32_16x16x32_bf16 (compute-bound; bf16 in / f32 accumulate).
// GEMMs are software-pipelined with double-buffered LDS so global tile loads
// overlap the WMMA stream instead of stalling on s_wait_loadcnt 0.
// ---------------------------------------------------------------------------

typedef __bf16 bf16_t;
typedef __attribute__((ext_vector_type(16))) __bf16 v16bf;
typedef __attribute__((ext_vector_type(8)))  float   v8f;

union FragBF { v16bf v; uint4 q[2]; };

#define T_TOK 8192
#define HDIM  1024
#define NEXP  8
#define IRDIM 4096
#define ISDIM 8192
#define CAPE  1280   // int(1.25 * 8192 / 8)

#define BM 128
#define BN 64
#define BK 32
#define LDS_A 40     // padded LDS row stride (bf16 elems) to dodge bank conflicts
#define LDS_B 40

__device__ __forceinline__ float fast_sigmoid(float g) {
  // v_exp_f32 + v_rcp_f32 instead of the IEEE divide expansion
  return __builtin_amdgcn_rcpf(1.f + __expf(-g));
}

// ------------------------- router (RMSNorm + softmax + top2) ----------------
__global__ __launch_bounds__(256) void moe_router(
    const float* __restrict__ x, const float* __restrict__ rw,
    bf16_t* __restrict__ xbf, int* __restrict__ tki, float* __restrict__ tkw)
{
  const int lane = threadIdx.x & 31;
  const int t = blockIdx.x * 8 + (threadIdx.x >> 5);   // one wave per token
  const float* xr = x + (long long)t * HDIM;
  float xs[32];
  float ss = 0.f;
#pragma unroll
  for (int j = 0; j < 32; ++j) {
    float v = xr[lane + 32 * j];
    xs[j] = v; ss += v * v;
    xbf[(long long)t * HDIM + lane + 32 * j] = (bf16_t)v;   // raw x in bf16 for experts
  }
#pragma unroll
  for (int o = 16; o; o >>= 1) ss += __shfl_xor(ss, o, 32);
  const float rstd = rsqrtf(ss * (1.f / HDIM) + 1.1920929e-07f);
  float logit[NEXP];
#pragma unroll
  for (int e = 0; e < NEXP; ++e) {
    const float* wr = rw + e * HDIM;
    float d = 0.f;
#pragma unroll
    for (int j = 0; j < 32; ++j) d += xs[j] * wr[lane + 32 * j];
#pragma unroll
    for (int o = 16; o; o >>= 1) d += __shfl_xor(d, o, 32);
    logit[e] = d * rstd;
  }
  if (lane == 0) {
    float mx = logit[0];
    for (int e = 1; e < NEXP; ++e) mx = fmaxf(mx, logit[e]);
    float p[NEXP];
    for (int e = 0; e < NEXP; ++e) p[e] = __expf(logit[e] - mx);
    int i0 = 0;
    for (int e = 1; e < NEXP; ++e) if (p[e] > p[i0]) i0 = e;     // strict > : stable
    int i1 = (i0 == 0) ? 1 : 0;
    for (int e = 0; e < NEXP; ++e) if (e != i0 && p[e] > p[i1]) i1 = e;
    float s2 = p[i0] + p[i1];
    tki[t * 2] = i0; tki[t * 2 + 1] = i1;
    tkw[t * 2] = p[i0] / s2; tkw[t * 2 + 1] = p[i1] / s2;
  }
}

// ---------------- per-expert capacity dispatch (stable token order) ---------
__global__ __launch_bounds__(1024) void moe_dispatch(
    const int* __restrict__ tki, const float* __restrict__ tkw,
    int* __restrict__ disp, float* __restrict__ dispw)
{
  const int e = blockIdx.x;
  __shared__ int wtot[32], wscan[32];
  __shared__ int base_s, ctot_s;
  const int tid = threadIdx.x, lane = tid & 31, wv = tid >> 5;

  for (int i = tid; i < CAPE; i += 1024) { disp[e * CAPE + i] = 0; dispw[e * CAPE + i] = 0.f; }
  if (tid == 0) base_s = 0;
  __syncthreads();

  for (int c = 0; c < T_TOK; c += 1024) {
    const int t = c + tid;
    const int i0 = tki[t * 2], i1 = tki[t * 2 + 1];
    const bool flag = (i0 == e) || (i1 == e);
    const float w = (i0 == e) ? tkw[t * 2] : ((i1 == e) ? tkw[t * 2 + 1] : 0.f);
    const unsigned int bm = (unsigned int)__ballot(flag);
    const int pre = __popc(bm & ((1u << lane) - 1u));
    if (lane == 0) wtot[wv] = __popc(bm);
    __syncthreads();
    if (tid == 0) {
      int s = 0;
      for (int i = 0; i < 32; ++i) { wscan[i] = s; s += wtot[i]; }
      ctot_s = s;
    }
    __syncthreads();
    const int pos = base_s + wscan[wv] + pre;
    if (flag && pos < CAPE) { disp[e * CAPE + pos] = t; dispw[e * CAPE + pos] = w; }
    __syncthreads();
    if (tid == 0) base_s += ctot_s;
    __syncthreads();
  }
}

// ------------- weight transpose + fp32 -> bf16  ([K][N] -> [N][K]) ----------
__global__ __launch_bounds__(256) void moe_tconv(
    const float* __restrict__ in, bf16_t* __restrict__ out,
    int Krows, int Ncols, long long strideIn, long long strideOut)
{
  __shared__ float tile[32 * 33];
  const int e = blockIdx.z;
  const float* I = in + (long long)e * strideIn;
  bf16_t* O = out + (long long)e * strideOut;
  const int k0 = blockIdx.y * 32, n0 = blockIdx.x * 32;
  for (int p = threadIdx.x; p < 1024; p += 256) {
    const int i = p >> 5, j = p & 31;
    tile[i * 33 + j] = I[(long long)(k0 + i) * Ncols + n0 + j];
  }
  __syncthreads();
  for (int p = threadIdx.x; p < 1024; p += 256) {
    const int n = p >> 5, k = p & 31;
    O[(long long)(n0 + n) * Krows + k0 + k] = (bf16_t)tile[k * 33 + n];
  }
}

// ----------------- fused gate/up GEMM + SiLU*up epilogue --------------------
// A: bf16 [*, Ktot] rows (optionally gathered via disp).  B1T/B2T: [N][K] bf16.
// Out: bf16 [M, Ntot].  Block tile 128x64, 8 waves (4x2), wave tile 32x32.
// Double-buffered LDS, register-staged global loads hidden under WMMAs.
__global__ __launch_bounds__(256) void moe_gateup(
    const bf16_t* __restrict__ A, const int* __restrict__ disp,
    const bf16_t* __restrict__ B1T, const bf16_t* __restrict__ B2T,
    bf16_t* __restrict__ Hout, int Ktot, int Ntot,
    long long strideB, long long strideH, int cap)
{
  __shared__ __align__(16) bf16_t As[2][BM * LDS_A];
  __shared__ __align__(16) bf16_t B1s[2][BN * LDS_B];
  __shared__ __align__(16) bf16_t B2s[2][BN * LDS_B];

  const int e = blockIdx.z;
  const bf16_t* b1 = B1T + (long long)e * strideB;
  const bf16_t* b2 = B2T + (long long)e * strideB;
  bf16_t* ho = Hout + (long long)e * strideH;
  const int* dp = disp ? (disp + e * cap) : nullptr;

  const int tid = threadIdx.x;
  const int lane = tid & 31, wid = tid >> 5;
  const int wm = wid >> 1, wn = wid & 1;
  const int tm = blockIdx.y * BM, tn = blockIdx.x * BN;
  const int kh = lane >> 4, lr = lane & 15;

  // ---- hoisted staging addresses (loop-invariant, incl. gather indices) ----
  const int rA0 = tid >> 2, gA = tid & 3;     // A: 2 tasks of (row, 8-elem group)
  const int rA1 = rA0 + 64;
  const long long grow0 = dp ? (long long)dp[tm + rA0] : (long long)(tm + rA0);
  const long long grow1 = dp ? (long long)dp[tm + rA1] : (long long)(tm + rA1);
  const char* aSrc0 = (const char*)(A + grow0 * Ktot) + gA * 16;
  const char* aSrc1 = (const char*)(A + grow1 * Ktot) + gA * 16;
  const int bN = tid >> 2, bG = tid & 3;      // B: 1 task per matrix
  const char* b1Src = (const char*)(b1 + (long long)(tn + bN) * Ktot) + bG * 16;
  const char* b2Src = (const char*)(b2 + (long long)(tn + bN) * Ktot) + bG * 16;
  const int aOff0 = rA0 * LDS_A + gA * 8;
  const int aOff1 = rA1 * LDS_A + gA * 8;
  const int bOff  = bN * LDS_B + bG * 8;

  v8f acc1[2][2] = {}, acc2[2][2] = {};

  // ---- prologue: stage tile 0 ----
  uint4 ra0 = *(const uint4*)aSrc0;
  uint4 ra1 = *(const uint4*)aSrc1;
  uint4 rb1 = *(const uint4*)b1Src;
  uint4 rb2 = *(const uint4*)b2Src;
  *(uint4*)(&As[0][aOff0]) = ra0;
  *(uint4*)(&As[0][aOff1]) = ra1;
  *(uint4*)(&B1s[0][bOff]) = rb1;
  *(uint4*)(&B2s[0][bOff]) = rb2;
  __syncthreads();

  int buf = 0;
  for (int k0 = 0; k0 < Ktot; k0 += BK) {
    const bool more = (k0 + BK) < Ktot;
    // (a) issue next tile's global loads first -- no wait here
    if (more) {
      const long long bo = (long long)(k0 + BK) * 2;   // bytes
      ra0 = *(const uint4*)(aSrc0 + bo);
      ra1 = *(const uint4*)(aSrc1 + bo);
      rb1 = *(const uint4*)(b1Src + bo);
      rb2 = *(const uint4*)(b2Src + bo);
      __builtin_prefetch(b1Src + bo + BK * 2, 0, 1);   // global_prefetch_b8 (L2 warm)
      __builtin_prefetch(b2Src + bo + BK * 2, 0, 1);
    }
    // (b) fragment ds_loads from current buffer
    const bf16_t* asb = As[buf];
    const bf16_t* b1b = B1s[buf];
    const bf16_t* b2b = B2s[buf];
    FragBF a[2], f1[2], f2[2];
#pragma unroll
    for (int im = 0; im < 2; ++im) {            // A 16x32: lane half -> 16 contiguous K
      const bf16_t* ap = asb + (wm * 32 + im * 16 + lr) * LDS_A + kh * 16;
      a[im].q[0] = *(const uint4*)ap;
      a[im].q[1] = *(const uint4*)(ap + 8);
    }
#pragma unroll
    for (int in_ = 0; in_ < 2; ++in_) {         // B 32x16: lane half -> 16 contiguous K
      const bf16_t* p1 = b1b + (wn * 32 + in_ * 16 + lr) * LDS_B + kh * 16;
      f1[in_].q[0] = *(const uint4*)p1;  f1[in_].q[1] = *(const uint4*)(p1 + 8);
      const bf16_t* p2 = b2b + (wn * 32 + in_ * 16 + lr) * LDS_B + kh * 16;
      f2[in_].q[0] = *(const uint4*)p2;  f2[in_].q[1] = *(const uint4*)(p2 + 8);
    }
    // (c) WMMAs cover the in-flight global loads
#pragma unroll
    for (int im = 0; im < 2; ++im)
#pragma unroll
      for (int in_ = 0; in_ < 2; ++in_) {
        acc1[im][in_] = __builtin_amdgcn_wmma_f32_16x16x32_bf16(
            false, a[im].v, false, f1[in_].v, (short)0, acc1[im][in_], false, false);
        acc2[im][in_] = __builtin_amdgcn_wmma_f32_16x16x32_bf16(
            false, a[im].v, false, f2[in_].v, (short)0, acc2[im][in_], false, false);
      }
    // (d) store prefetched regs into the other buffer (loadcnt wait lands here)
    if (more) {
      const int nb = buf ^ 1;
      *(uint4*)(&As[nb][aOff0]) = ra0;
      *(uint4*)(&As[nb][aOff1]) = ra1;
      *(uint4*)(&B1s[nb][bOff]) = rb1;
      *(uint4*)(&B2s[nb][bOff]) = rb2;
    }
    // (e) single barrier per K step
    __syncthreads();
    buf ^= 1;
  }

  #pragma unroll
  for (int im = 0; im < 2; ++im)                // C/D: M = r + 8*(lane>=16), N = lane&15
#pragma unroll
    for (int in_ = 0; in_ < 2; ++in_) {
      const int mb = tm + wm * 32 + im * 16 + kh * 8;
      const int nn = tn + wn * 32 + in_ * 16 + lr;
#pragma unroll
      for (int r = 0; r < 8; ++r) {
        const float g = acc1[im][in_][r];
        const float u = acc2[im][in_][r];
        const float h = g * fast_sigmoid(g) * u;        // silu(g) * u
        ho[(long long)(mb + r) * Ntot + nn] = (bf16_t)h;
      }
    }
}

// --------------- down GEMM: store (shared) or weighted scatter-add ----------
__global__ __launch_bounds__(256) void moe_down(
    const bf16_t* __restrict__ Abase, const bf16_t* __restrict__ BT,
    float* __restrict__ Out, const int* __restrict__ disp,
    const float* __restrict__ dispw, int Ktot, int Ntot,
    long long strideA, long long strideBexp, int cap, float scale)
{
  __shared__ __align__(16) bf16_t As[2][BM * LDS_A];
  __shared__ __align__(16) bf16_t Bs[2][BN * LDS_B];

  const int e = blockIdx.z;
  const bf16_t* A = Abase + (long long)e * strideA;
  const bf16_t* Bt = BT + (long long)e * strideBexp;
  const int* dp = disp ? (disp + e * cap) : nullptr;
  const float* dw = dispw ? (dispw + e * cap) : nullptr;

  const int tid = threadIdx.x;
  const int lane = tid & 31, wid = tid >> 5;
  const int wm = wid >> 1, wn = wid & 1;
  const int tm = blockIdx.y * BM, tn = blockIdx.x * BN;
  const int kh = lane >> 4, lr = lane & 15;

  const int rA0 = tid >> 2, gA = tid & 3;
  const int rA1 = rA0 + 64;
  const char* aSrc0 = (const char*)(A + (long long)(tm + rA0) * Ktot) + gA * 16;
  const char* aSrc1 = (const char*)(A + (long long)(tm + rA1) * Ktot) + gA * 16;
  const int bN = tid >> 2, bG = tid & 3;
  const char* bSrc = (const char*)(Bt + (long long)(tn + bN) * Ktot) + bG * 16;
  const int aOff0 = rA0 * LDS_A + gA * 8;
  const int aOff1 = rA1 * LDS_A + gA * 8;
  const int bOff  = bN * LDS_B + bG * 8;

  v8f acc[2][2] = {};

  uint4 ra0 = *(const uint4*)aSrc0;
  uint4 ra1 = *(const uint4*)aSrc1;
  uint4 rb  = *(const uint4*)bSrc;
  *(uint4*)(&As[0][aOff0]) = ra0;
  *(uint4*)(&As[0][aOff1]) = ra1;
  *(uint4*)(&Bs[0][bOff])  = rb;
  __syncthreads();

  int buf = 0;
  for (int k0 = 0; k0 < Ktot; k0 += BK) {
    const bool more = (k0 + BK) < Ktot;
    if (more) {
      const long long bo = (long long)(k0 + BK) * 2;
      ra0 = *(const uint4*)(aSrc0 + bo);
      ra1 = *(const uint4*)(aSrc1 + bo);
      rb  = *(const uint4*)(bSrc + bo);
      __builtin_prefetch(bSrc + bo + BK * 2, 0, 1);
    }
    const bf16_t* asb = As[buf];
    const bf16_t* bsb = Bs[buf];
    FragBF a[2], f[2];
#pragma unroll
    for (int im = 0; im < 2; ++im) {
      const bf16_t* ap = asb + (wm * 32 + im * 16 + lr) * LDS_A + kh * 16;
      a[im].q[0] = *(const uint4*)ap;
      a[im].q[1] = *(const uint4*)(ap + 8);
    }
#pragma unroll
    for (int in_ = 0; in_ < 2; ++in_) {
      const bf16_t* p1 = bsb + (wn * 32 + in_ * 16 + lr) * LDS_B + kh * 16;
      f[in_].q[0] = *(const uint4*)p1;  f[in_].q[1] = *(const uint4*)(p1 + 8);
    }
#pragma unroll
    for (int im = 0; im < 2; ++im)
#pragma unroll
      for (int in_ = 0; in_ < 2; ++in_)
        acc[im][in_] = __builtin_amdgcn_wmma_f32_16x16x32_bf16(
            false, a[im].v, false, f[in_].v, (short)0, acc[im][in_], false, false);
    if (more) {
      const int nb = buf ^ 1;
      *(uint4*)(&As[nb][aOff0]) = ra0;
      *(uint4*)(&As[nb][aOff1]) = ra1;
      *(uint4*)(&Bs[nb][bOff])  = rb;
    }
    __syncthreads();
    buf ^= 1;
  }

#pragma unroll
  for (int im = 0; im < 2; ++im)
#pragma unroll
    for (int in_ = 0; in_ < 2; ++in_) {
      const int mb = tm + wm * 32 + im * 16 + kh * 8;
      const int nn = tn + wn * 32 + in_ * 16 + lr;
#pragma unroll
      for (int r = 0; r < 8; ++r) {
        const float v = acc[im][in_][r];
        const int m = mb + r;
        if (dp) {
          const int tok = dp[m];
          const float w = dw[m];                        // 0 on filler/dropped slots
          atomicAdd(&Out[(long long)tok * Ntot + nn], scale * w * v);
        } else {
          Out[(long long)m * Ntot + nn] = v;
        }
      }
    }
}

// ---------------------------------------------------------------------------
extern "C" void kernel_launch(void* const* d_in, const int* in_sizes, int n_in,
                              void* d_out, int out_size, void* d_ws, size_t ws_size,
                              hipStream_t stream)
{
  (void)in_sizes; (void)n_in; (void)out_size; (void)ws_size;
  const float* x  = (const float*)d_in[0];
  const float* rw = (const float*)d_in[1];
  const float* rg = (const float*)d_in[2];
  const float* ru = (const float*)d_in[3];
  const float* rd = (const float*)d_in[4];
  const float* sg = (const float*)d_in[5];
  const float* su = (const float*)d_in[6];
  const float* sd = (const float*)d_in[7];
  float* out = (float*)d_out;

  char* p = (char*)d_ws;
  auto take = [&](size_t bytes) -> char* {
    char* r = p; p += (bytes + 255) & ~(size_t)255; return r;
  };
  bf16_t* xbf = (bf16_t*)take((size_t)T_TOK * HDIM * 2);
  bf16_t* sgT = (bf16_t*)take((size_t)ISDIM * HDIM * 2);
  bf16_t* suT = (bf16_t*)take((size_t)ISDIM * HDIM * 2);
  bf16_t* sdT = (bf16_t*)take((size_t)HDIM * ISDIM * 2);
  bf16_t* rgT = (bf16_t*)take((size_t)NEXP * IRDIM * HDIM * 2);
  bf16_t* ruT = (bf16_t*)take((size_t)NEXP * IRDIM * HDIM * 2);
  bf16_t* rdT = (bf16_t*)take((size_t)NEXP * HDIM * IRDIM * 2);
  bf16_t* hs  = (bf16_t*)take((size_t)T_TOK * ISDIM * 2);
  bf16_t* hr  = (bf16_t*)take((size_t)NEXP * CAPE * IRDIM * 2);
  int*    tki = (int*)  take((size_t)T_TOK * 2 * 4);
  float*  tkw = (float*)take((size_t)T_TOK * 2 * 4);
  int*    dsp = (int*)  take((size_t)NEXP * CAPE * 4);
  float*  dsw = (float*)take((size_t)NEXP * CAPE * 4);

  // 1) router + x->bf16
  moe_router<<<T_TOK / 8, 256, 0, stream>>>(x, rw, xbf, tki, tkw);
  // 2) capacity dispatch per expert
  moe_dispatch<<<NEXP, 1024, 0, stream>>>(tki, tkw, dsp, dsw);
  // 3) weight transpose+convert: [K][N] fp32 -> [N][K] bf16
  { dim3 g(ISDIM / 32, HDIM / 32, 1);     moe_tconv<<<g, 256, 0, stream>>>(sg, sgT, HDIM, ISDIM, 0, 0); }
  { dim3 g(ISDIM / 32, HDIM / 32, 1);     moe_tconv<<<g, 256, 0, stream>>>(su, suT, HDIM, ISDIM, 0, 0); }
  { dim3 g(HDIM / 32, ISDIM / 32, 1);     moe_tconv<<<g, 256, 0, stream>>>(sd, sdT, ISDIM, HDIM, 0, 0); }
  { dim3 g(IRDIM / 32, HDIM / 32, NEXP);  moe_tconv<<<g, 256, 0, stream>>>(rg, rgT, HDIM, IRDIM, (long long)HDIM * IRDIM, (long long)IRDIM * HDIM); }
  { dim3 g(IRDIM / 32, HDIM / 32, NEXP);  moe_tconv<<<g, 256, 0, stream>>>(ru, ruT, HDIM, IRDIM, (long long)HDIM * IRDIM, (long long)IRDIM * HDIM); }
  { dim3 g(HDIM / 32, IRDIM / 32, NEXP);  moe_tconv<<<g, 256, 0, stream>>>(rd, rdT, IRDIM, HDIM, (long long)IRDIM * HDIM, (long long)HDIM * IRDIM); }
  // 4) shared expert gate/up (fused): hs = silu(x@sg) * (x@su)
  { dim3 g(ISDIM / BN, T_TOK / BM, 1);
    moe_gateup<<<g, 256, 0, stream>>>(xbf, nullptr, sgT, suT, hs, HDIM, ISDIM, 0, 0, CAPE); }
  // 5) routed experts gate/up, rows gathered via dispatch
  { dim3 g(IRDIM / BN, CAPE / BM, NEXP);
    moe_gateup<<<g, 256, 0, stream>>>(xbf, dsp, rgT, ruT, hr, HDIM, IRDIM,
                                      (long long)IRDIM * HDIM, (long long)CAPE * IRDIM, CAPE); }
  // 6) shared down: out = hs @ sd  (overwrites d_out)
  { dim3 g(HDIM / BN, T_TOK / BM, 1);
    moe_down<<<g, 256, 0, stream>>>(hs, sdT, out, nullptr, nullptr, ISDIM, HDIM, 0, 0, CAPE, 1.0f); }
  // 7) routed down + weighted scatter-add (SCALE = 1.0)
  { dim3 g(HDIM / BN, CAPE / BM, NEXP);
    moe_down<<<g, 256, 0, stream>>>(hr, rdT, out, dsp, dsw, IRDIM, HDIM,
                                    (long long)CAPE * IRDIM, (long long)HDIM * IRDIM, CAPE, 1.0f); }
}
